// BiDAF_9758165696824
// MI455X (gfx1250) — matrix operand order repaired and checked by reading, not compiled, not run on hardware
//
#include <hip/hip_runtime.h>
#include <hip/hip_bf16.h>

// ---------------------------------------------------------------------------
// CDNA5 (gfx1250) BiDAF implementation.
// All dense linear algebra runs through v_wmma_f32_16x16x32_f16 (wave32).
// GEMM: each wave computes a 32x64 tile (2x4 WMMA tiles, register-blocked).
// LSTM recurrences: persistent workgroups, h/c in LDS/regs, Whh from L2,
// A-fragments (broadcast h) hoisted into registers once per time step,
// two independent gate-tile accumulation chains per wave, and
// global_prefetch of the next step's xg row under the WMMA chain.
// ---------------------------------------------------------------------------

typedef __attribute__((ext_vector_type(16))) _Float16 v16h;
typedef __attribute__((ext_vector_type(8)))  _Float16 v8h;
typedef __attribute__((ext_vector_type(8)))  float    v8f;

#define DEVINL static __device__ __forceinline__

// Load A fragment (16x32 f16) per ISA layout:
//   lanes 0-15: M=lane,    K = k..k+7  and k+16..k+23
//   lanes16-31: M=lane-16, K = k+8..k+15 and k+24..k+31
DEVINL v16h load_a_frag(const _Float16* __restrict__ A, int lda, int m0, int k, int lane) {
  int m   = m0 + (lane & 15);
  int klo = k + ((lane & 16) ? 8 : 0);
  const _Float16* p = A + (size_t)m * lda + klo;
  v8h a = *(const v8h*)p;
  v8h b = *(const v8h*)(p + 16);
  v16h r;
#pragma unroll
  for (int i = 0; i < 8; ++i) { r[i] = a[i]; r[i + 8] = b[i]; }
  return r;
}

// Load B fragment (32x16 f16) from row-major weight W[N,K]:
//   lane n = lane&15 ; lanes 0-15 hold K=k..k+7 and k+16..k+23, lanes 16-31 hold K=k+8..k+15 and k+24..k+31
DEVINL v16h load_b_frag(const _Float16* __restrict__ W, int ldb, int n0, int k, int lane) {
  int n  = n0 + (lane & 15);
  int kb = k + ((lane & 16) ? 8 : 0);
  const _Float16* p = W + (size_t)n * ldb + kb;
  v8h a = *(const v8h*)p;
  v8h b = *(const v8h*)(p + 16);
  v16h r;
#pragma unroll
  for (int i = 0; i < 8; ++i) { r[i] = a[i]; r[i + 8] = b[i]; }
  return r;
}

// ---------------------------------------------------------------------------
// WMMA GEMM:  Y[M,N] (f32) = A[M,K] (f16) @ B[N,K]^T (f16) + bias
// One wave per 32x64 output tile (2 M-tiles x 4 N-tiles, 8 accumulators).
// M mult of 32, N mult of 64, K mult of 32 (caller pads).
// ---------------------------------------------------------------------------
__global__ void wmma_gemm_bias_kernel(const _Float16* __restrict__ A,
                                      const _Float16* __restrict__ B,
                                      const float* __restrict__ bias0,
                                      float* __restrict__ Y,
                                      int Kpad, int lda, int ldb, int ldy) {
  int lane = threadIdx.x;        // 32 threads = 1 wave
  int n0 = blockIdx.x << 6;
  int m0 = blockIdx.y << 5;
  v8f accA[4] = {};
  v8f accB[4] = {};
  for (int k = 0; k < Kpad; k += 32) {
    v16h a0 = load_a_frag(A, lda, m0, k, lane);
    v16h a1 = load_a_frag(A, lda, m0 + 16, k, lane);
#pragma unroll
    for (int t = 0; t < 4; ++t) {
      v16h b = load_b_frag(B, ldb, n0 + (t << 4), k, lane);
      accA[t] = __builtin_amdgcn_wmma_f32_16x16x32_f16(false, a0, false, b,
                                                       (short)0, accA[t], false, false);
      accB[t] = __builtin_amdgcn_wmma_f32_16x16x32_f16(false, a1, false, b,
                                                       (short)0, accB[t], false, false);
    }
  }
  int lnib = lane & 15;
  int roff = (lane & 16) ? 8 : 0;
#pragma unroll
  for (int t = 0; t < 4; ++t) {
    int n = n0 + (t << 4) + lnib;
    float bv = bias0 ? bias0[n] : 0.f;
#pragma unroll
    for (int r = 0; r < 8; ++r) {
      Y[(size_t)(m0 + roff + r) * ldy + n]      = accA[t][r] + bv;
      Y[(size_t)(m0 + 16 + roff + r) * ldy + n] = accB[t][r] + bv;
    }
  }
}

// ---------------------------------------------------------------------------
// Persistent LSTM recurrence. One 256-thread block (8 waves) per job.
// gates[t] = xg[t] + h @ W^T via WMMA with h broadcast into all 16 A-rows
// (row 0 of D is the matvec).  A fragments hoisted per step; B streams from L2;
// two gate tiles are accumulated concurrently to deepen issue parallelism.
// ---------------------------------------------------------------------------
struct LstmJob {
  const float* xg;       // [T, ldxg] input-side gate preactivations
  int ldxg;
  const _Float16* W;     // [G, KPAD] recurrent weights (f16, K padded)
  float* hout;           // [T, H]
  int T;
  int rev;               // 0 = forward, 1 = reverse traversal
};
struct LstmJobs {
  LstmJob j[4];
  int H, G;
};

template <int KPAD>
__global__ void lstm_recurrent_kernel(LstmJobs jobs) {
  constexpr int NK = KPAD / 32;
  LstmJob jb = jobs.j[blockIdx.x];
  const int H = jobs.H, G = jobs.G;

  __shared__ __align__(16) _Float16 hsh[KPAD];
  __shared__ float gsh[1024];

  int tid  = threadIdx.x;
  int lane = tid & 31;
  int wave = tid >> 5;

  if (tid < KPAD) hsh[tid] = (_Float16)0.f;
  float c = 0.f;
  __syncthreads();

  int ntile = G >> 4;

  for (int step = 0; step < jb.T; ++step) {
    int t = jb.rev ? (jb.T - 1 - step) : step;

    // Prefetch next step's xg row into cache under this step's WMMA chain.
    if (step + 1 < jb.T) {
      int tn = jb.rev ? (jb.T - 2 - step) : (step + 1);
      const float* nrow = jb.xg + (size_t)tn * jb.ldxg;
      int poff = tid << 5;               // 128B per thread
      if (poff < G) __builtin_prefetch(nrow + poff, 0, 0);
    }

    // Hoist A fragments (all 16 rows = h) for every K chunk, once per step.
    v16h afr[NK];
#pragma unroll
    for (int kk = 0; kk < NK; ++kk) {
      int klo = (kk << 5) + ((lane & 16) ? 8 : 0);
      v8h alo = *(const v8h*)&hsh[klo];
      v8h ahi = *(const v8h*)&hsh[klo + 16];
#pragma unroll
      for (int i = 0; i < 8; ++i) { afr[kk][i] = alo[i]; afr[kk][i + 8] = ahi[i]; }
    }

    const float* xrow = jb.xg + (size_t)t * jb.ldxg;
    // Two tiles per iteration -> two independent WMMA accumulation chains.
    for (int tile = wave; tile < ntile; tile += 16) {
      int n0 = tile << 4;
      int tile1 = tile + 8;
      bool two = tile1 < ntile;          // uniform across the wave
      int n1 = tile1 << 4;
      v8f acc0 = {};
      v8f acc1 = {};
#pragma unroll
      for (int kk = 0; kk < NK; ++kk) {
        int k = kk << 5;
        v16h b0 = load_b_frag(jb.W, KPAD, n0, k, lane);
        acc0 = __builtin_amdgcn_wmma_f32_16x16x32_f16(false, afr[kk], false, b0,
                                                      (short)0, acc0, false, false);
        if (two) {
          v16h b1 = load_b_frag(jb.W, KPAD, n1, k, lane);
          acc1 = __builtin_amdgcn_wmma_f32_16x16x32_f16(false, afr[kk], false, b1,
                                                        (short)0, acc1, false, false);
        }
      }
      if (lane < 16) {                   // row M=0 lives in acc[0] of lanes 0..15
        gsh[n0 + lane] = acc0[0] + xrow[n0 + lane];
        if (two) gsh[n1 + lane] = acc1[0] + xrow[n1 + lane];
      }
    }
    __syncthreads();

    if (tid < H) {
      float ig = gsh[tid], fg = gsh[H + tid], gg = gsh[2 * H + tid], og = gsh[3 * H + tid];
      float si = 1.f / (1.f + __expf(-ig));
      float sf = 1.f / (1.f + __expf(-fg));
      float so = 1.f / (1.f + __expf(-og));
      c = sf * c + si * tanhf(gg);
      float h = so * tanhf(c);
      hsh[tid] = (_Float16)h;
      jb.hout[(size_t)t * H + tid] = h;
    }
    __syncthreads();
  }
}

// ------------------------- small utility kernels ---------------------------

__global__ void cvt_pad_f16_kernel(const float* __restrict__ src, _Float16* __restrict__ dst,
                                   int rows, int cols, int ldd, int rowsPad) {
  long i = (long)blockIdx.x * blockDim.x + threadIdx.x;
  long total = (long)rowsPad * ldd;
  if (i >= total) return;
  int r = (int)(i / ldd), c = (int)(i % ldd);
  float v = (r < rows && c < cols) ? src[(size_t)r * cols + c] : 0.f;
  dst[i] = (_Float16)v;
}

__global__ void addvec_pad_kernel(const float* a, const float* b, float* o, int n, int npad) {
  int i = blockIdx.x * blockDim.x + threadIdx.x;
  if (i >= npad) return;
  o[i] = (i < n) ? a[i] + b[i] : 0.f;
}

__global__ void gather_char_kernel(const int* __restrict__ idx, const float* __restrict__ table,
                                   _Float16* __restrict__ X, int n) {
  int i = blockIdx.x * blockDim.x + threadIdx.x;
  if (i >= n * 64) return;
  int r = i >> 6, c = i & 63;
  X[i] = (_Float16)table[(size_t)idx[r] * 64 + c];
}

// X[r, 0:300]=word emb, X[r,300:400]=mean of 8 char-LSTM states, X[r,400:416]=0
__global__ void build_enc_input_kernel(const int* __restrict__ tok, const float* __restrict__ embW,
                                       const float* __restrict__ charh,
                                       _Float16* __restrict__ X, int n) {
  int i = blockIdx.x * blockDim.x + threadIdx.x;
  if (i >= n * 416) return;
  int r = i / 416, c = i % 416;
  float v = 0.f;
  if (c < 300) {
    v = embW[(size_t)tok[r] * 300 + c];
  } else if (c < 400) {
    int ch = c - 300;
    float s = 0.f;
    for (int w = 0; w < 8; ++w) s += charh[(size_t)(r * 8 + w) * 100 + ch];
    v = s * 0.125f;
  }
  X[i] = (_Float16)v;
}

__global__ void concat_dirs_kernel(const float* __restrict__ hs, float* __restrict__ O,
                                   _Float16* __restrict__ O16, int T, int H) {
  int i = blockIdx.x * blockDim.x + threadIdx.x;
  if (i >= T * 2 * H) return;
  int t = i / (2 * H), c = i % (2 * H);
  float v = (c < H) ? hs[(size_t)t * H + c] : hs[(size_t)(T + t) * H + (c - H)];
  O[i] = v;
  if (O16) O16[i] = (_Float16)v;
}

__global__ void scale_rows_f16_kernel(const float* __restrict__ X, const float* __restrict__ w,
                                      _Float16* __restrict__ Y, long total, int cols) {
  long i = (long)blockIdx.x * blockDim.x + threadIdx.x;
  if (i >= total) return;
  int c = (int)(i % cols);
  Y[i] = (_Float16)(X[i] * w[c]);
}

__global__ void transpose_f16_kernel(const float* __restrict__ X, _Float16* __restrict__ Y,
                                     int rows, int cols) {
  int i = blockIdx.x * blockDim.x + threadIdx.x;
  if (i >= rows * cols) return;
  int r = i / cols, c = i % cols;
  Y[(size_t)c * rows + r] = (_Float16)X[i];
}

__global__ void dot_rows_kernel(const float* __restrict__ X, const float* __restrict__ w,
                                float* __restrict__ out, int rows, int cols) {
  int r = blockIdx.x * blockDim.x + threadIdx.x;
  if (r >= rows) return;
  float a = 0.f;
  for (int c = 0; c < cols; ++c) a += X[(size_t)r * cols + c] * w[c];
  out[r] = a;
}

// S[t,j] = Sraw + sC[t] + sQ[j]; write row softmax (f16) and row max
__global__ void att_post_kernel(const float* __restrict__ Sraw, const float* __restrict__ sC,
                                const float* __restrict__ sQ, _Float16* __restrict__ P16,
                                float* __restrict__ mrow) {
  int t = blockIdx.x * blockDim.x + threadIdx.x;
  if (t >= 1024) return;
  float s[64], mx = -1e30f;
  for (int j = 0; j < 64; ++j) {
    float v = Sraw[t * 64 + j] + sC[t] + sQ[j];
    s[j] = v;
    mx = fmaxf(mx, v);
  }
  float sum = 0.f;
  for (int j = 0; j < 64; ++j) { float e = __expf(s[j] - mx); s[j] = e; sum += e; }
  float inv = 1.f / sum;
  for (int j = 0; j < 64; ++j) P16[t * 64 + j] = (_Float16)(s[j] * inv);
  mrow[t] = mx;
}

__global__ void htilde_kernel(const float* __restrict__ mrow, const float* __restrict__ C,
                              float* __restrict__ ht) {
  int d = blockIdx.x * blockDim.x + threadIdx.x;
  if (d >= 512) return;
  float a = 0.f;
  for (int t = 0; t < 1024; ++t) a += mrow[t] * C[(size_t)t * 512 + d];
  ht[d] = a;
}

__global__ void build_g_kernel(const float* __restrict__ C, const float* __restrict__ c2q,
                               const float* __restrict__ ht, float* __restrict__ G,
                               _Float16* __restrict__ Gh) {
  int i = blockIdx.x * blockDim.x + threadIdx.x;
  if (i >= 1024 * 512) return;
  int t = i >> 9, d = i & 511;
  float cv = C[i], av = c2q[i];
  float g0 = cv, g1 = av, g2 = cv * av, g3 = cv * ht[d];
  size_t base = (size_t)t * 2048;
  G[base + d] = g0;        G[base + 512 + d]  = g1;
  G[base + 1024 + d] = g2; G[base + 1536 + d] = g3;
  Gh[base + d] = (_Float16)g0;        Gh[base + 512 + d]  = (_Float16)g1;
  Gh[base + 1024 + d] = (_Float16)g2; Gh[base + 1536 + d] = (_Float16)g3;
}

// logit[t] = G[t,:2048].w[:2048] + X2[t,:512].w[2048:2560]
__global__ void logits_kernel(const float* __restrict__ G, const float* __restrict__ X2,
                              const float* __restrict__ w, float* __restrict__ out) {
  int t = blockIdx.x * blockDim.x + threadIdx.x;
  if (t >= 1024) return;
  float a = 0.f;
  for (int c = 0; c < 2048; ++c) a += G[(size_t)t * 2048 + c] * w[c];
  for (int c = 0; c < 512; ++c)  a += X2[(size_t)t * 512 + c] * w[2048 + c];
  out[t] = a;
}

__global__ void softmax1024_kernel(const float* __restrict__ in, float* __restrict__ out) {
  __shared__ float red[1024];
  int tid = threadIdx.x;
  float v = in[tid];
  red[tid] = v;
  __syncthreads();
  for (int s = 512; s > 0; s >>= 1) {
    if (tid < s) red[tid] = fmaxf(red[tid], red[tid + s]);
    __syncthreads();
  }
  float mx = red[0];
  __syncthreads();
  float e = __expf(v - mx);
  red[tid] = e;
  __syncthreads();
  for (int s = 512; s > 0; s >>= 1) {
    if (tid < s) red[tid] += red[tid + s];
    __syncthreads();
  }
  out[tid] = e / red[0];
}

// ---------------------------------------------------------------------------
// Host orchestration
// ---------------------------------------------------------------------------
struct Bump {
  char* base;
  size_t off;
  void* take(size_t bytes) {
    void* p = base + off;
    off = (off + bytes + 255) & ~(size_t)255;
    return p;
  }
};

static inline int cdiv(long a, int b) { return (int)((a + b - 1) / b); }

extern "C" void kernel_launch(void* const* d_in, const int* in_sizes, int n_in,
                              void* d_out, int out_size, void* d_ws, size_t ws_size,
                              hipStream_t stream) {
  (void)in_sizes; (void)n_in; (void)out_size; (void)ws_size;
  const int T = 1024, J = 64, WL = 8;
  const int H = 256, G = 1024;        // word LSTM hidden / gates
  const int CH = 100, GC = 400;       // char LSTM hidden / gates
  const int GCP = 448;                // char gates padded to mult of 64 (GEMM N)
  const int KE = 416;                 // E+CH=400 padded to 32
  const int KHC = 128;                // CH padded to 32
  const int NC = T * WL, NQ = J * WL; // 8192 / 512 chars
  const int Dm = 512;                 // bidirectional dim

  const int*   context    = (const int*)d_in[0];
  const int*   ctx_chars  = (const int*)d_in[1];
  const int*   query      = (const int*)d_in[2];
  const int*   qry_chars  = (const int*)d_in[3];
  const float* emb_W      = (const float*)d_in[4];
  const float* char_emb_W = (const float*)d_in[5];
  const float* char_Wih   = (const float*)d_in[6];
  const float* char_Whh   = (const float*)d_in[7];
  const float* char_bih   = (const float*)d_in[8];
  const float* char_bhh   = (const float*)d_in[9];
  const float* ctx_Wih    = (const float*)d_in[10];
  const float* ctx_Whh    = (const float*)d_in[11];
  const float* ctx_bih    = (const float*)d_in[12];
  const float* ctx_bhh    = (const float*)d_in[13];
  const float* qry_Wih    = (const float*)d_in[14];
  const float* qry_Whh    = (const float*)d_in[15];
  const float* qry_bih    = (const float*)d_in[16];
  const float* qry_bhh    = (const float*)d_in[17];
  const float* sim_w      = (const float*)d_in[18];
  const float* mod1_Wih   = (const float*)d_in[19];
  const float* mod1_Whh   = (const float*)d_in[20];
  const float* mod1_bih   = (const float*)d_in[21];
  const float* mod1_bhh   = (const float*)d_in[22];
  const float* mod2_Wih   = (const float*)d_in[23];
  const float* mod2_Whh   = (const float*)d_in[24];
  const float* mod2_bih   = (const float*)d_in[25];
  const float* mod2_bhh   = (const float*)d_in[26];
  const float* start_w    = (const float*)d_in[27];
  const float* end_Wih    = (const float*)d_in[28];
  const float* end_Whh    = (const float*)d_in[29];
  const float* end_bih    = (const float*)d_in[30];
  const float* end_bhh    = (const float*)d_in[31];
  const float* end_w      = (const float*)d_in[32];

  Bump ws{(char*)d_ws, 0};
  typedef _Float16 h16;

  // ---- f16 weight copies (K padded to mult of 32; char gate rows to 448) ----
  h16* charWih16 = (h16*)ws.take((size_t)GCP * 64 * 2);
  h16* charWhh16 = (h16*)ws.take((size_t)GC * KHC * 2);
  h16* ctxWih16  = (h16*)ws.take((size_t)2048 * KE * 2);
  h16* ctxWhh16  = (h16*)ws.take((size_t)2048 * 256 * 2);
  h16* qryWih16  = (h16*)ws.take((size_t)2048 * KE * 2);
  h16* qryWhh16  = (h16*)ws.take((size_t)2048 * 256 * 2);
  h16* m1Wih16   = (h16*)ws.take((size_t)2048 * 2048 * 2);
  h16* m1Whh16   = (h16*)ws.take((size_t)2048 * 256 * 2);
  h16* m2Wih16   = (h16*)ws.take((size_t)2048 * 512 * 2);
  h16* m2Whh16   = (h16*)ws.take((size_t)2048 * 256 * 2);
  h16* enWih16   = (h16*)ws.take((size_t)2048 * 512 * 2);
  h16* enWhh16   = (h16*)ws.take((size_t)2048 * 256 * 2);
  float* bchar = (float*)ws.take(GCP * 4);
  float* bctx  = (float*)ws.take(2048 * 4);
  float* bqry  = (float*)ws.take(2048 * 4);
  float* bm1   = (float*)ws.take(2048 * 4);
  float* bm2   = (float*)ws.take(2048 * 4);
  float* ben   = (float*)ws.take(2048 * 4);

  // ---- activations ----
  h16*   XcC   = (h16*)ws.take((size_t)NC * 64 * 2);
  h16*   XcQ   = (h16*)ws.take((size_t)NQ * 64 * 2);
  float* xgcC  = (float*)ws.take((size_t)NC * GCP * 4);
  float* xgcQ  = (float*)ws.take((size_t)NQ * GCP * 4);
  float* chhC  = (float*)ws.take((size_t)NC * CH * 4);
  float* chhQ  = (float*)ws.take((size_t)NQ * CH * 4);
  h16*   XeC   = (h16*)ws.take((size_t)T * KE * 2);
  h16*   XeQ   = (h16*)ws.take((size_t)J * KE * 2);
  float* xgBuf = (float*)ws.take((size_t)T * 2048 * 4);   // reused ctx/mod1/mod2/end
  float* xgQry = (float*)ws.take((size_t)J * 2048 * 4);
  float* hsBuf = (float*)ws.take((size_t)2 * T * H * 4);  // reused per long bilstm
  float* hsQ   = (float*)ws.take((size_t)2 * J * H * 4);
  float* Cf    = (float*)ws.take((size_t)T * Dm * 4);
  float* Qf    = (float*)ws.take((size_t)J * Dm * 4);
  h16*   Q16   = (h16*)ws.take((size_t)J * Dm * 2);
  h16*   Qt16  = (h16*)ws.take((size_t)Dm * J * 2);
  h16*   Cm16  = (h16*)ws.take((size_t)T * Dm * 2);
  float* sC    = (float*)ws.take(T * 4);
  float* sQ    = (float*)ws.take(J * 4);
  float* Sraw  = (float*)ws.take((size_t)T * J * 4);
  h16*   P16   = (h16*)ws.take((size_t)T * J * 2);
  float* mrow  = (float*)ws.take(T * 4);
  float* ht    = (float*)ws.take(Dm * 4);
  float* c2q   = (float*)ws.take((size_t)T * Dm * 4);
  float* Gf    = (float*)ws.take((size_t)T * 2048 * 4);
  h16*   Gh16  = (h16*)ws.take((size_t)T * 2048 * 2);
  float* M1f   = (float*)ws.take((size_t)T * Dm * 4);
  h16*   M1h   = (h16*)ws.take((size_t)T * Dm * 2);
  float* Mof   = (float*)ws.take((size_t)T * Dm * 4);
  h16*   Moh   = (h16*)ws.take((size_t)T * Dm * 2);
  float* Ehf   = (float*)ws.take((size_t)T * Dm * 4);
  float* logit = (float*)ws.take(T * 4);

  const int B = 256;
#define EW(total) cdiv((long)(total), B), B, 0, stream

  auto mkjob = [](const float* xg, int ldxg, const h16* W, float* hout, int T_, int rev) {
    LstmJob j; j.xg = xg; j.ldxg = ldxg; j.W = W; j.hout = hout; j.T = T_; j.rev = rev;
    return j;
  };

  // ---- convert weights ----
  cvt_pad_f16_kernel<<<EW((long)GCP * 64)>>>(char_Wih, charWih16, GC, 64, 64, GCP);
  cvt_pad_f16_kernel<<<EW((long)GC * KHC)>>>(char_Whh, charWhh16, GC, 100, KHC, GC);
  cvt_pad_f16_kernel<<<EW((long)2048 * KE)>>>(ctx_Wih, ctxWih16, 2048, 400, KE, 2048);
  cvt_pad_f16_kernel<<<EW((long)2048 * 256)>>>(ctx_Whh, ctxWhh16, 2048, 256, 256, 2048);
  cvt_pad_f16_kernel<<<EW((long)2048 * KE)>>>(qry_Wih, qryWih16, 2048, 400, KE, 2048);
  cvt_pad_f16_kernel<<<EW((long)2048 * 256)>>>(qry_Whh, qryWhh16, 2048, 256, 256, 2048);
  cvt_pad_f16_kernel<<<EW((long)2048 * 2048)>>>(mod1_Wih, m1Wih16, 2048, 2048, 2048, 2048);
  cvt_pad_f16_kernel<<<EW((long)2048 * 256)>>>(mod1_Whh, m1Whh16, 2048, 256, 256, 2048);
  cvt_pad_f16_kernel<<<EW((long)2048 * 512)>>>(mod2_Wih, m2Wih16, 2048, 512, 512, 2048);
  cvt_pad_f16_kernel<<<EW((long)2048 * 256)>>>(mod2_Whh, m2Whh16, 2048, 256, 256, 2048);
  cvt_pad_f16_kernel<<<EW((long)2048 * 512)>>>(end_Wih, enWih16, 2048, 512, 512, 2048);
  cvt_pad_f16_kernel<<<EW((long)2048 * 256)>>>(end_Whh, enWhh16, 2048, 256, 256, 2048);
  addvec_pad_kernel<<<EW(GCP)>>>(char_bih, char_bhh, bchar, GC, GCP);
  addvec_pad_kernel<<<EW(2048)>>>(ctx_bih, ctx_bhh, bctx, 2048, 2048);
  addvec_pad_kernel<<<EW(2048)>>>(qry_bih, qry_bhh, bqry, 2048, 2048);
  addvec_pad_kernel<<<EW(2048)>>>(mod1_bih, mod1_bhh, bm1, 2048, 2048);
  addvec_pad_kernel<<<EW(2048)>>>(mod2_bih, mod2_bhh, bm2, 2048, 2048);
  addvec_pad_kernel<<<EW(2048)>>>(end_bih, end_bhh, ben, 2048, 2048);

  // ---- char pipeline (ctx + qry sequences run concurrently, grid=2) ----
  gather_char_kernel<<<EW((long)NC * 64)>>>(ctx_chars, char_emb_W, XcC, NC);
  gather_char_kernel<<<EW((long)NQ * 64)>>>(qry_chars, char_emb_W, XcQ, NQ);
  wmma_gemm_bias_kernel<<<dim3(GCP / 64, NC / 32), 32, 0, stream>>>(
      XcC, charWih16, bchar, xgcC, 64, 64, 64, GCP);
  wmma_gemm_bias_kernel<<<dim3(GCP / 64, NQ / 32), 32, 0, stream>>>(
      XcQ, charWih16, bchar, xgcQ, 64, 64, 64, GCP);
  {
    LstmJobs jc;
    jc.j[0] = mkjob(xgcC, GCP, charWhh16, chhC, NC, 0);
    jc.j[1] = mkjob(xgcQ, GCP, charWhh16, chhQ, NQ, 0);
    jc.j[2] = jc.j[0]; jc.j[3] = jc.j[0];
    jc.H = CH; jc.G = GC;
    lstm_recurrent_kernel<128><<<2, 256, 0, stream>>>(jc);
  }

  // ---- ctx + qry encoders (4 directions run concurrently, grid=4) ----
  build_enc_input_kernel<<<EW((long)T * KE)>>>(context, emb_W, chhC, XeC, T);
  build_enc_input_kernel<<<EW((long)J * KE)>>>(query, emb_W, chhQ, XeQ, J);
  wmma_gemm_bias_kernel<<<dim3(2048 / 64, T / 32), 32, 0, stream>>>(
      XeC, ctxWih16, bctx, xgBuf, KE, KE, KE, 2048);
  wmma_gemm_bias_kernel<<<dim3(2048 / 64, J / 32), 32, 0, stream>>>(
      XeQ, qryWih16, bqry, xgQry, KE, KE, KE, 2048);
  {
    LstmJobs je;
    je.j[0] = mkjob(xgBuf,        2048, ctxWhh16,             hsBuf,           T, 0);
    je.j[1] = mkjob(xgBuf + G,    2048, ctxWhh16 + G * 256,   hsBuf + T * H,   T, 1);
    je.j[2] = mkjob(xgQry,        2048, qryWhh16,             hsQ,             J, 0);
    je.j[3] = mkjob(xgQry + G,    2048, qryWhh16 + G * 256,   hsQ + J * H,     J, 1);
    je.H = H; je.G = G;
    lstm_recurrent_kernel<256><<<4, 256, 0, stream>>>(je);
  }
  concat_dirs_kernel<<<EW((long)T * Dm)>>>(hsBuf, Cf, (h16*)nullptr, T, H);
  concat_dirs_kernel<<<EW((long)J * Dm)>>>(hsQ, Qf, Q16, J, H);

  // ---- attention ----
  scale_rows_f16_kernel<<<EW((long)T * Dm)>>>(Cf, sim_w + 1024, Cm16, (long)T * Dm, Dm);
  dot_rows_kernel<<<EW(T)>>>(Cf, sim_w, sC, T, Dm);
  dot_rows_kernel<<<EW(J)>>>(Qf, sim_w + 512, sQ, J, Dm);
  wmma_gemm_bias_kernel<<<dim3(J / 64, T / 32), 32, 0, stream>>>(
      Cm16, Q16, nullptr, Sraw, Dm, Dm, Dm, J);
  att_post_kernel<<<EW(T)>>>(Sraw, sC, sQ, P16, mrow);
  transpose_f16_kernel<<<EW((long)J * Dm)>>>(Qf, Qt16, J, Dm);
  wmma_gemm_bias_kernel<<<dim3(Dm / 64, T / 32), 32, 0, stream>>>(
      P16, Qt16, nullptr, c2q, J, J, J, Dm);
  htilde_kernel<<<EW(Dm)>>>(mrow, Cf, ht);
  build_g_kernel<<<EW((long)T * Dm)>>>(Cf, c2q, ht, Gf, Gh16);

  // ---- modeling layer 1 ----
  wmma_gemm_bias_kernel<<<dim3(2048 / 64, T / 32), 32, 0, stream>>>(
      Gh16, m1Wih16, bm1, xgBuf, 2048, 2048, 2048, 2048);
  {
    LstmJobs jm;
    jm.j[0] = mkjob(xgBuf,     2048, m1Whh16,           hsBuf,         T, 0);
    jm.j[1] = mkjob(xgBuf + G, 2048, m1Whh16 + G * 256, hsBuf + T * H, T, 1);
    jm.j[2] = jm.j[0]; jm.j[3] = jm.j[0];
    jm.H = H; jm.G = G;
    lstm_recurrent_kernel<256><<<2, 256, 0, stream>>>(jm);
  }
  concat_dirs_kernel<<<EW((long)T * Dm)>>>(hsBuf, M1f, M1h, T, H);

  // ---- modeling layer 2 ----
  wmma_gemm_bias_kernel<<<dim3(2048 / 64, T / 32), 32, 0, stream>>>(
      M1h, m2Wih16, bm2, xgBuf, 512, 512, 512, 2048);
  {
    LstmJobs jm;
    jm.j[0] = mkjob(xgBuf,     2048, m2Whh16,           hsBuf,         T, 0);
    jm.j[1] = mkjob(xgBuf + G, 2048, m2Whh16 + G * 256, hsBuf + T * H, T, 1);
    jm.j[2] = jm.j[0]; jm.j[3] = jm.j[0];
    jm.H = H; jm.G = G;
    lstm_recurrent_kernel<256><<<2, 256, 0, stream>>>(jm);
  }
  concat_dirs_kernel<<<EW((long)T * Dm)>>>(hsBuf, Mof, Moh, T, H);

  // ---- start head ----
  logits_kernel<<<EW(T)>>>(Gf, Mof, start_w, logit);
  softmax1024_kernel<<<1, 1024, 0, stream>>>(logit, (float*)d_out);

  // ---- end head ----
  wmma_gemm_bias_kernel<<<dim3(2048 / 64, T / 32), 32, 0, stream>>>(
      Moh, enWih16, ben, xgBuf, 512, 512, 512, 2048);
  {
    LstmJobs jm;
    jm.j[0] = mkjob(xgBuf,     2048, enWhh16,           hsBuf,         T, 0);
    jm.j[1] = mkjob(xgBuf + G, 2048, enWhh16 + G * 256, hsBuf + T * H, T, 1);
    jm.j[2] = jm.j[0]; jm.j[3] = jm.j[0];
    jm.H = H; jm.G = G;
    lstm_recurrent_kernel<256><<<2, 256, 0, stream>>>(jm);
  }
  concat_dirs_kernel<<<EW((long)T * Dm)>>>(hsBuf, Ehf, (h16*)nullptr, T, H);
  logits_kernel<<<EW(T)>>>(Gf, Ehf, end_w, logit);
  softmax1024_kernel<<<1, 1024, 0, stream>>>(logit, (float*)d_out + 1024);
#undef EW
}
